// Encoder_24249385353447
// MI455X (gfx1250) — matrix operand order repaired and checked by reading, not compiled, not run on hardware
//
#include <hip/hip_runtime.h>
#include <hip/hip_bf16.h>

// Problem dims (compile-time constants from the reference)
#define B_ 256
#define T_ 512
#define KOBS 8
#define E_ 32
#define H_ 128
#define G_ 512           // 4*H
#define KC 160           // H + E  (combined K for [h|x] @ [W_hh|W_ih]^T)
#define KP 168           // padded K stride in f16 elems: 336B, 16B-aligned, good LDS banking
#define GP 513           // padded gate row stride (f32)
#define BS 16            // batch rows per workgroup (WMMA M)
#define NWG (B_ / BS)    // 16 persistent workgroups
#define NTILES (G_ / 16) // 32 N-tiles of the 512-wide gate output
#define THREADS 256
#define WAVES (THREADS / 32)
#define TILES_PER_WAVE (NTILES / WAVES) // 4

typedef __attribute__((ext_vector_type(16))) _Float16 v16h;
typedef __attribute__((ext_vector_type(8)))  _Float16 v8h;
typedef __attribute__((ext_vector_type(8)))  float    v8f;

// Load a 16-f16 WMMA fragment as two 16B LDS vectors.
__device__ __forceinline__ v16h ld_frag(const _Float16* p0, const _Float16* p1) {
    v8h lo = *(const v8h*)p0;
    v8h hi = *(const v8h*)p1;
    return __builtin_shufflevector(lo, hi, 0, 1, 2, 3, 4, 5, 6, 7,
                                           8, 9, 10, 11, 12, 13, 14, 15);
}

__device__ __forceinline__ float sigmoidf_(float x) {
    return 1.0f / (1.0f + __expf(-x));
}

// ---------------------------------------------------------------------------
// Kernel 1: embedding. act path = single gather; obs path = masked gather-sum.
// Writes the `embedded` output region (also consumed by the LSTM kernel).
// ---------------------------------------------------------------------------
__global__ void embed_kernel(const int* __restrict__ action_ids,   // [B,T]
                             const int* __restrict__ obs_ids,      // [B,T,K]
                             const int* __restrict__ is_action,    // [B,T]
                             const int* __restrict__ obs_mask,     // [B,T,K]
                             const float* __restrict__ atab,       // [V,E]
                             const float* __restrict__ otab,       // [V,E]
                             float* __restrict__ emb)              // [B,T,E]
{
    int idx = blockIdx.x * blockDim.x + threadIdx.x;
    if (idx >= B_ * T_ * E_) return;
    int e  = idx & (E_ - 1);
    int bt = idx >> 5;
    float v;
    if (is_action[bt]) {
        v = atab[action_ids[bt] * E_ + e];
    } else {
        float s = 0.0f;
#pragma unroll
        for (int k = 0; k < KOBS; ++k) {
            s += (float)obs_mask[bt * KOBS + k] * otab[obs_ids[bt * KOBS + k] * E_ + e];
        }
        v = s;
    }
    emb[idx] = v;
}

// ---------------------------------------------------------------------------
// Kernel 2: persistent LSTM. One workgroup per 16-row batch slice; combined
// weights [W_hh | W_ih] held resident in LDS (f16) for the whole T loop.
// Per step (2 barriers, double-buffered A):
//   gates(16x512) = WMMA([h|x](16x160) x Wc^T(160x512))           -> LDS
//   elementwise cell (+bias), masked output store, stage h & x_{t+1}
//   into the other A buffer.
// ---------------------------------------------------------------------------
__global__ void __launch_bounds__(THREADS)
lstm_kernel(const float* __restrict__ emb,    // [B,T,E] (written by embed_kernel)
            const int*   __restrict__ lens,   // [B]
            const float* __restrict__ W_ih,   // [G,E]
            const float* __restrict__ W_hh,   // [G,H]
            const float* __restrict__ b_ih,   // [G]
            const float* __restrict__ b_hh,   // [G]
            float* __restrict__ outs,         // [B,T,H]
            float* __restrict__ h_n,          // [B,H]
            float* __restrict__ c_n)          // [B,H]
{
    __shared__ __align__(16) _Float16 Wc[G_][KP];     // 168 KB: combined weights, f16
    __shared__ __align__(16) _Float16 Aq[2][BS][KP];  // ping-pong A rows: [h | x_t]
    __shared__ float Gt[BS][GP];                      // gate accumulators (f32)
    __shared__ float Hs[BS][H_];
    __shared__ float Cs[BS][H_];
    __shared__ float Bias[G_];
    __shared__ int   Len[BS];

    const int tid  = threadIdx.x;
    const int wave = tid >> 5;
    const int lane = tid & 31;
    const int b0   = blockIdx.x * BS;

    // ---- one-time init: fp32 weights -> f16 LDS (Wc[n][k] = k<H ? W_hh : W_ih)
    for (int i = tid; i < G_ * KC; i += THREADS) {
        int n = i / KC, k = i % KC;
        float w = (k < H_) ? W_hh[n * H_ + k] : W_ih[n * E_ + (k - H_)];
        Wc[n][k] = (_Float16)w;
    }
    for (int n = tid; n < G_; n += THREADS) Bias[n] = b_ih[n] + b_hh[n];
    for (int i = tid; i < BS * H_; i += THREADS) {
        int b = i >> 7, j = i & (H_ - 1);
        Hs[b][j] = 0.0f;
        Cs[b][j] = 0.0f;
        Aq[0][b][j] = (_Float16)0.0f;   // h_0 = 0 in A buffer 0
    }
    // stage x_0 into A buffer 0
    for (int i = tid; i < BS * E_; i += THREADS) {
        int b = i >> 5, e = i & (E_ - 1);
        Aq[0][b][H_ + e] = (_Float16)emb[((size_t)(b0 + b) * T_ + 0) * E_ + e];
    }
    if (tid < BS) Len[tid] = lens[b0 + tid];
    __syncthreads();

    const int m_a = lane & 15;   // A row / B column this lane serves
    const int hi  = lane >> 4;   // lane half selects K sub-range per ISA layout

    for (int t = 0; t < T_; ++t) {
        const int cur = t & 1, nxt = cur ^ 1;

        // -- phase A: gates = A @ Wc^T via v_wmma_f32_16x16x32_f16 (zero-init acc)
        v16h af[5];
#pragma unroll
        for (int q = 0; q < 5; ++q) {
            // A layout (16-bit, 16x32): lanes 0-15 hold K {0..7,16..23},
            // lanes 16-31 hold K {8..15,24..31} of chunk q.
            int kb = q * 32 + hi * 8;
            af[q] = ld_frag(&Aq[cur][m_a][kb], &Aq[cur][m_a][kb + 16]);
        }
#pragma unroll
        for (int ti = 0; ti < TILES_PER_WAVE; ++ti) {
            int ntile = wave * TILES_PER_WAVE + ti;
            int ng    = ntile * 16 + m_a;      // global gate column for this lane
            v8f acc   = {};                    // folds to WMMA inline SRC2=0
#pragma unroll
            for (int q = 0; q < 5; ++q) {
                // B layout (16-bit, 32x16): lane n holds K {0..7,16..23} of column n,
                // lane n+16 holds K {8..15,24..31}. B[k][n] = Wc[n][k] -> row reads.
                int kb  = q * 32 + hi * 8;
                v16h bf = ld_frag(&Wc[ng][kb], &Wc[ng][kb + 16]);
                acc = __builtin_amdgcn_wmma_f32_16x16x32_f16(
                          false, af[q], false, bf, (short)0, acc, false, false);
            }
            // C/D layout: vgpr r, lane half -> row r + hi*8, col = lane&15
#pragma unroll
            for (int r = 0; r < 8; ++r)
                Gt[r + hi * 8][ng] = acc[r];
        }
        __syncthreads();

        // -- phase B: elementwise LSTM cell with bias (torch gate order i,f,g,o),
        //    masked output store, stage h_{t+1} and x_{t+1} into the other A buffer.
#pragma unroll
        for (int s = 0; s < (BS * H_) / THREADS; ++s) {
            int idx = tid + s * THREADS;
            int b = idx >> 7, j = idx & (H_ - 1);
            float ig = sigmoidf_(Gt[b][j]          + Bias[j]);
            float fg = sigmoidf_(Gt[b][H_ + j]     + Bias[H_ + j]);
            float gg = tanhf(    Gt[b][2 * H_ + j] + Bias[2 * H_ + j]);
            float og = sigmoidf_(Gt[b][3 * H_ + j] + Bias[3 * H_ + j]);
            float c  = Cs[b][j];
            float h  = Hs[b][j];
            float cn = fg * c + ig * gg;
            float hn = og * tanhf(cn);
            float m  = (t < Len[b]) ? 1.0f : 0.0f;
            outs[((size_t)(b0 + b) * T_ + t) * H_ + j] = hn * m;
            float h2 = m * hn + (1.0f - m) * h;
            float c2 = m * cn + (1.0f - m) * c;
            Hs[b][j] = h2;
            Cs[b][j] = c2;
            Aq[nxt][b][j] = (_Float16)h2;   // h rows for next step's A matrix
        }
        if (t + 1 < T_) {
            for (int i = tid; i < BS * E_; i += THREADS) {
                int b = i >> 5, e = i & (E_ - 1);
                Aq[nxt][b][H_ + e] =
                    (_Float16)emb[((size_t)(b0 + b) * T_ + (t + 1)) * E_ + e];
            }
            if (t + 2 < T_ && tid < BS)
                __builtin_prefetch(&emb[((size_t)(b0 + tid) * T_ + (t + 2)) * E_], 0, 0);
        }
        __syncthreads();
    }

    // final states
    for (int i = tid; i < BS * H_; i += THREADS) {
        int b = i >> 7, j = i & (H_ - 1);
        h_n[(size_t)(b0 + b) * H_ + j] = Hs[b][j];
        c_n[(size_t)(b0 + b) * H_ + j] = Cs[b][j];
    }
}

// ---------------------------------------------------------------------------
extern "C" void kernel_launch(void* const* d_in, const int* in_sizes, int n_in,
                              void* d_out, int out_size, void* d_ws, size_t ws_size,
                              hipStream_t stream) {
    const int*   action_ids = (const int*)d_in[0];
    const int*   obs_ids    = (const int*)d_in[1];
    const int*   is_action  = (const int*)d_in[2];
    const int*   obs_mask   = (const int*)d_in[3];
    const int*   lens       = (const int*)d_in[4];
    const float* atab       = (const float*)d_in[5];
    const float* otab       = (const float*)d_in[6];
    const float* W_ih       = (const float*)d_in[7];
    const float* W_hh       = (const float*)d_in[8];
    const float* b_ih       = (const float*)d_in[9];
    const float* b_hh       = (const float*)d_in[10];

    float* out   = (float*)d_out;
    float* outs  = out;                                     // [B,T,H]
    float* h_n   = outs + (size_t)B_ * T_ * H_;             // [1,B,H]
    float* c_n   = h_n + (size_t)B_ * H_;                   // [1,B,H]
    float* embp  = c_n + (size_t)B_ * H_;                   // [B,T,E]

    int n = B_ * T_ * E_;
    embed_kernel<<<(n + 255) / 256, 256, 0, stream>>>(
        action_ids, obs_ids, is_action, obs_mask, atab, otab, embp);

    lstm_kernel<<<NWG, THREADS, 0, stream>>>(
        embp, lens, W_ih, W_hh, b_ih, b_hh, outs, h_n, c_n);
}
